// FusedMoEWithLoRA_79800492359939
// MI455X (gfx1250) — matrix operand order repaired and checked by reading, not compiled, not run on hardware
//
#include <hip/hip_runtime.h>

// ---------------- problem constants ----------------
#define T_TOK  16384
#define H_DIM  2048
#define I_DIM  1024
#define E_NUM  8
#define K_TOP  2
#define R_LORA 16

#define ROWS_CAP  33792          // T*K + E*128 worst-case padded rows
#define MAX_TILES 272            // >= sum_e ceil(cnt_e/128) <= 264
#define TILE_M    128
#define KB        64             // K-block staged in LDS
#define LDA       72             // LDS row stride in halves (64 + 8 pad, keeps 16B align)

typedef __attribute__((ext_vector_type(16))) __bf16 v16bf;
typedef __attribute__((ext_vector_type(8)))  __bf16 v8bf;
typedef __attribute__((ext_vector_type(4)))  __bf16 v4bf;
typedef __attribute__((ext_vector_type(8)))  float  v8f;

__device__ __forceinline__ v16bf cat8(v8bf lo, v8bf hi) {
    return __builtin_shufflevector(lo, hi, 0,1,2,3,4,5,6,7,8,9,10,11,12,13,14,15);
}
__device__ __forceinline__ v8f wmma_bf16(v16bf a, v16bf b, v8f c) {
    // (neg_a, A, neg_b, B, c_mod, C, reuse_a, reuse_b)
    return __builtin_amdgcn_wmma_f32_16x16x32_bf16(false, a, false, b, (short)0, c, false, false);
}

// -------- CDNA5 async global->LDS copy (ASYNCcnt) with guarded fallback --------
#if defined(__HIP_DEVICE_COMPILE__) && __has_builtin(__builtin_amdgcn_global_load_async_to_lds_b128)
#define USE_ASYNC 1
#else
#define USE_ASYNC 0
#if defined(__HIP_DEVICE_COMPILE__)
#warning "global_load_async_to_lds_b128 builtin unavailable: using synchronous LDS staging"
#endif
#endif

#if USE_ASYNC
typedef int i32x4 __attribute__((vector_size(16)));
typedef __attribute__((address_space(1))) i32x4* gp128_t;   // global int32x4*
typedef __attribute__((address_space(3))) i32x4* lp128_t;   // LDS    int32x4*
__device__ __forceinline__ void async_cp_row64(const __bf16* g, __bf16* l) {
    __builtin_amdgcn_global_load_async_to_lds_b128((gp128_t)(g +  0), (lp128_t)(l +  0), 0, 0);
    __builtin_amdgcn_global_load_async_to_lds_b128((gp128_t)(g +  8), (lp128_t)(l +  8), 0, 0);
    __builtin_amdgcn_global_load_async_to_lds_b128((gp128_t)(g + 16), (lp128_t)(l + 16), 0, 0);
    __builtin_amdgcn_global_load_async_to_lds_b128((gp128_t)(g + 24), (lp128_t)(l + 24), 0, 0);
}
__device__ __forceinline__ void wait_async0() {
#if __has_builtin(__builtin_amdgcn_s_wait_asynccnt)
    __builtin_amdgcn_s_wait_asynccnt(0);
#else
    asm volatile("s_wait_asynccnt 0" ::: "memory");
#endif
}
#endif

// meta layout (ints): [0..7]=counts, [8..15]=cursor, [16]=ntiles,
// [32 .. 32+MAX_TILES) = tile_expert, [32+MAX_TILES .. 32+2*MAX_TILES) = tile_row0

// ---------------- setup kernels ----------------
__global__ void k_zero_out(float4* out, long n4) {
    long i = (long)blockIdx.x * blockDim.x + threadIdx.x;
    if (i < n4) out[i] = make_float4(0.f, 0.f, 0.f, 0.f);
}

__global__ void k_convert_x(const float* __restrict__ x, __bf16* __restrict__ xb, long n4) {
    long i = (long)blockIdx.x * blockDim.x + threadIdx.x;
    if (i >= n4) return;
    float4 v = ((const float4*)x)[i];
    v4bf o;
    o.x = (__bf16)v.x; o.y = (__bf16)v.y; o.z = (__bf16)v.z; o.w = (__bf16)v.w;
    ((v4bf*)xb)[i] = o;
}

// w13m[e][n][k] = bf16( w13[e][n][k] + sum_r b1[e][n][r] * a1[e][r][k] )
__global__ void k_merge_w13(const float* __restrict__ w13,
                            const float* __restrict__ la,   // [E,R,H]
                            const float* __restrict__ lb,   // [E,2I,R]
                            __bf16* __restrict__ w13m) {
    long idx = (long)blockIdx.x * blockDim.x + threadIdx.x;
    const long total = (long)E_NUM * 2 * I_DIM * H_DIM;
    if (idx >= total) return;
    int e = (int)(idx / ((long)2 * I_DIM * H_DIM));
    long rem = idx - (long)e * 2 * I_DIM * H_DIM;
    int n = (int)(rem / H_DIM);
    int k = (int)(rem - (long)n * H_DIM);
    const float* b = lb + ((long)e * 2 * I_DIM + n) * R_LORA;
    const float* a = la + (long)e * R_LORA * H_DIM + k;
    float acc = w13[idx];
#pragma unroll
    for (int r = 0; r < R_LORA; ++r) acc += b[r] * a[(long)r * H_DIM];
    w13m[idx] = (__bf16)acc;
}

// w2m[e][n][k] = bf16( w2[e][n][k] + sum_r b2[e][n][r] * a2[e][r][k] )
__global__ void k_merge_w2(const float* __restrict__ w2,
                           const float* __restrict__ la,    // [E,R,I]
                           const float* __restrict__ lb,    // [E,H,R]
                           __bf16* __restrict__ w2m) {
    long idx = (long)blockIdx.x * blockDim.x + threadIdx.x;
    const long total = (long)E_NUM * H_DIM * I_DIM;
    if (idx >= total) return;
    int e = (int)(idx / ((long)H_DIM * I_DIM));
    long rem = idx - (long)e * H_DIM * I_DIM;
    int n = (int)(rem / I_DIM);
    int k = (int)(rem - (long)n * I_DIM);
    const float* b = lb + ((long)e * H_DIM + n) * R_LORA;
    const float* a = la + (long)e * R_LORA * I_DIM + k;
    float acc = w2[idx];
#pragma unroll
    for (int r = 0; r < R_LORA; ++r) acc += b[r] * a[(long)r * I_DIM];
    w2m[idx] = (__bf16)acc;
}

// ---------------- routing kernels ----------------
__global__ void k_fill(int* g_token, float* g_weight, int* meta) {
    int i = blockIdx.x * blockDim.x + threadIdx.x;
    if (i < ROWS_CAP) { g_token[i] = -1; g_weight[i] = 0.f; }
    if (i < 32) meta[i] = 0;
}

__global__ void k_count(const int* __restrict__ ids, int* meta) {
    int t = blockIdx.x * blockDim.x + threadIdx.x;
    if (t < T_TOK) {
        atomicAdd(&meta[ids[t * K_TOP + 0]], 1);
        atomicAdd(&meta[ids[t * K_TOP + 1]], 1);
    }
}

__global__ void k_scan(int* meta) {
    if (blockIdx.x != 0 || threadIdx.x != 0) return;
    int row = 0, tiles = 0;
    for (int e = 0; e < E_NUM; ++e) {
        int cnt = meta[e];
        meta[8 + e] = row;                      // scatter cursor = segment base
        int nt = (cnt + TILE_M - 1) / TILE_M;
        for (int t = 0; t < nt; ++t) {
            meta[32 + tiles] = e;
            meta[32 + MAX_TILES + tiles] = row + t * TILE_M;
            ++tiles;
        }
        row += nt * TILE_M;
    }
    meta[16] = tiles;
}

__global__ void k_scatter(const int* __restrict__ ids, const float* __restrict__ w,
                          int* g_token, float* g_weight, int* meta) {
    int t = blockIdx.x * blockDim.x + threadIdx.x;
    if (t >= T_TOK) return;
    for (int k = 0; k < K_TOP; ++k) {
        int e = ids[t * K_TOP + k];
        int pos = atomicAdd(&meta[8 + e], 1);
        g_token[pos] = t;
        g_weight[pos] = w[t * K_TOP + k];
    }
}

// ---------------- GEMM1: act = silu(Xg @ Wg.T) * (Xg @ Wu.T) ----------------
// grid = (MAX_TILES, I/128), block = 256 (8 waves: 4 M x 2 N)
__global__ __launch_bounds__(256)
void k_gemm1(const __bf16* __restrict__ xb,    // [T, H] bf16
             const __bf16* __restrict__ w13m,  // [E, 2I, H] bf16 (merged LoRA)
             __bf16* __restrict__ act,         // [ROWS_CAP, I] bf16
             const int* __restrict__ g_token,
             const int* __restrict__ meta) {
    const int tile = blockIdx.x;
    if (tile >= meta[16]) return;
    const int e     = meta[32 + tile];
    const int row0  = meta[32 + MAX_TILES + tile];
    const int ibase = blockIdx.y * 128;        // gate cols [ibase, ibase+128); up = +I

    __shared__ __align__(16) __bf16 As[2][TILE_M * LDA];

    const int tid  = threadIdx.x;
    const int lane = tid & 31;
    const int wid  = tid >> 5;
    const int wm   = wid & 3;                   // rows wm*32
    const int wn   = wid >> 2;                  // cols wn*64

    // per-thread A staging: row lrow, 32 halves at lkoff
    const int lrow  = tid >> 1;
    const int lkoff = (tid & 1) * 32;
    const int tok   = g_token[row0 + lrow];
    const __bf16* xsrc = xb + (long)(tok < 0 ? 0 : tok) * H_DIM + lkoff;

    v8f accg[2][4], accu[2][4];
#pragma unroll
    for (int mt = 0; mt < 2; ++mt)
#pragma unroll
        for (int nt = 0; nt < 4; ++nt) { accg[mt][nt] = (v8f)0.f; accu[mt][nt] = (v8f)0.f; }

    const int arow = lane & 15;
    const int ak0  = (lane >> 4) ? 16 : 0;      // A: 16 consecutive K at ak0
    const int bk0  = (lane >> 4) ? 16 : 0;      // B: 16 consecutive K at bk0
    const long wbase = (long)e * 2 * I_DIM * H_DIM;

    // compute one staged K-block from LDS buffer Ab
    auto compute = [&](const __bf16* Ab, int k0) {
#pragma unroll
        for (int kc = 0; kc < 2; ++kc) {
            v16bf afrag[2];
#pragma unroll
            for (int mt = 0; mt < 2; ++mt) {
                const __bf16* p = &Ab[(wm * 32 + mt * 16 + arow) * LDA + kc * 32 + ak0];
                afrag[mt] = cat8(*(const v8bf*)p, *(const v8bf*)(p + 8));
            }
            // 8 B fragments (gate/up x 4 subtiles), 3-deep rotating prefetch queue
            v16bf bq[8];
#pragma unroll
            for (int pre = 0; pre < 3; ++pre) {
                const int nt = pre >> 1, gu = pre & 1;
                const __bf16* p = w13m + wbase +
                    (long)(ibase + wn * 64 + nt * 16 + (lane & 15) + (gu ? I_DIM : 0)) * H_DIM +
                    k0 + kc * 32 + bk0;
                bq[pre] = cat8(*(const v8bf*)p, *(const v8bf*)(p + 8));
            }
#pragma unroll
            for (int s = 0; s < 8; ++s) {
                if (s + 3 < 8) {
                    const int q = s + 3, nt = q >> 1, gu = q & 1;
                    const __bf16* p = w13m + wbase +
                        (long)(ibase + wn * 64 + nt * 16 + (lane & 15) + (gu ? I_DIM : 0)) * H_DIM +
                        k0 + kc * 32 + bk0;
                    bq[q] = cat8(*(const v8bf*)p, *(const v8bf*)(p + 8));
                }
                const int nt = s >> 1;
                if (s & 1) {
                    accu[0][nt] = wmma_bf16(afrag[0], bq[s], accu[0][nt]);
                    accu[1][nt] = wmma_bf16(afrag[1], bq[s], accu[1][nt]);
                } else {
                    accg[0][nt] = wmma_bf16(afrag[0], bq[s], accg[0][nt]);
                    accg[1][nt] = wmma_bf16(afrag[1], bq[s], accg[1][nt]);
                }
            }
        }
    };

#if USE_ASYNC
    // prologue: pre-zero padding rows in both buffers; async-stage first K-block
    if (tok < 0) {
        uint4 z = {0u, 0u, 0u, 0u};
        uint4* d0 = (uint4*)&As[0][lrow * LDA + lkoff];
        uint4* d1 = (uint4*)&As[1][lrow * LDA + lkoff];
        d0[0] = z; d0[1] = z; d0[2] = z; d0[3] = z;
        d1[0] = z; d1[1] = z; d1[2] = z; d1[3] = z;
    } else {
        async_cp_row64(xsrc, &As[0][lrow * LDA + lkoff]);
    }
    int cur = 0;
    for (int k0 = 0; k0 < H_DIM; k0 += KB) {
        wait_async0();
        __syncthreads();
        if (k0 + KB < H_DIM) {
            if (tok >= 0) async_cp_row64(xsrc + k0 + KB, &As[cur ^ 1][lrow * LDA + lkoff]);
#pragma unroll
            for (int nt = 0; nt < 4; ++nt) {   // prefetch next B block rows
                const long ncol = ibase + wn * 64 + nt * 16 + (lane & 15);
                __builtin_prefetch(w13m + wbase + ncol * H_DIM + k0 + KB, 0, 0);
                __builtin_prefetch(w13m + wbase + (ncol + I_DIM) * H_DIM + k0 + KB, 0, 0);
            }
        }
        compute(As[cur], k0);
        cur ^= 1;
    }
#else
    for (int k0 = 0; k0 < H_DIM; k0 += KB) {
        __syncthreads();
        uint4* d = (uint4*)&As[0][lrow * LDA + lkoff];
        if (tok >= 0) {
            const uint4* s = (const uint4*)(xsrc + k0);
            d[0] = s[0]; d[1] = s[1]; d[2] = s[2]; d[3] = s[3];
        } else {
            uint4 z = {0u, 0u, 0u, 0u};
            d[0] = z; d[1] = z; d[2] = z; d[3] = z;
        }
        __syncthreads();
        compute(As[0], k0);
    }
#endif

    // epilogue: act = silu(gate) * up, stored bf16 (C/D layout: row = j + (lane>=16)*8)
#pragma unroll
    for (int mt = 0; mt < 2; ++mt) {
#pragma unroll
        for (int nt = 0; nt < 4; ++nt) {
            const int col = ibase + wn * 64 + nt * 16 + (lane & 15);
#pragma unroll
            for (int j = 0; j < 8; ++j) {
                float g = accg[mt][nt][j];
                float u = accu[mt][nt][j];
                float s = g / (1.0f + __expf(-g));
                int mloc = (lane < 16) ? j : (8 + j);
                int row = row0 + wm * 32 + mt * 16 + mloc;
                act[(long)row * I_DIM + col] = (__bf16)(s * u);
            }
        }
    }
}

// ---------------- GEMM2: out[token] += rw * (act @ W2.T) ----------------
// grid = (MAX_TILES, H/128), block = 256
__global__ __launch_bounds__(256)
void k_gemm2(const __bf16* __restrict__ act,   // [ROWS_CAP, I] bf16
             const __bf16* __restrict__ w2m,   // [E, H, I] bf16 (merged LoRA)
             float* __restrict__ out,          // [T, H] f32
             const int* __restrict__ g_token,
             const float* __restrict__ g_weight,
             const int* __restrict__ meta) {
    const int tile = blockIdx.x;
    if (tile >= meta[16]) return;
    const int e     = meta[32 + tile];
    const int row0  = meta[32 + MAX_TILES + tile];
    const int nbase = blockIdx.y * 128;

    __shared__ __align__(16) __bf16 As[2][TILE_M * LDA];

    const int tid  = threadIdx.x;
    const int lane = tid & 31;
    const int wid  = tid >> 5;
    const int wm   = wid & 3;
    const int wn   = wid >> 2;

    const int lrow  = tid >> 1;
    const int lkoff = (tid & 1) * 32;
    const __bf16* asrc = act + (long)(row0 + lrow) * I_DIM + lkoff;

    v8f acc[2][4];
#pragma unroll
    for (int mt = 0; mt < 2; ++mt)
#pragma unroll
        for (int nt = 0; nt < 4; ++nt) acc[mt][nt] = (v8f)0.f;

    const int arow = lane & 15;
    const int ak0  = (lane >> 4) ? 16 : 0;
    const int bk0  = (lane >> 4) ? 16 : 0;
    const long wbase = (long)e * H_DIM * I_DIM;

    auto compute = [&](const __bf16* Ab, int k0) {
#pragma unroll
        for (int kc = 0; kc < 2; ++kc) {
            v16bf afrag[2];
#pragma unroll
            for (int mt = 0; mt < 2; ++mt) {
                const __bf16* p = &Ab[(wm * 32 + mt * 16 + arow) * LDA + kc * 32 + ak0];
                afrag[mt] = cat8(*(const v8bf*)p, *(const v8bf*)(p + 8));
            }
            v16bf bq[4];
#pragma unroll
            for (int pre = 0; pre < 2; ++pre) {
                const __bf16* p = w2m + wbase +
                    (long)(nbase + wn * 64 + pre * 16 + (lane & 15)) * I_DIM + k0 + kc * 32 + bk0;
                bq[pre] = cat8(*(const v8bf*)p, *(const v8bf*)(p + 8));
            }
#pragma unroll
            for (int nt = 0; nt < 4; ++nt) {
                if (nt + 2 < 4) {
                    const int q = nt + 2;
                    const __bf16* p = w2m + wbase +
                        (long)(nbase + wn * 64 + q * 16 + (lane & 15)) * I_DIM + k0 + kc * 32 + bk0;
                    bq[q] = cat8(*(const v8bf*)p, *(const v8bf*)(p + 8));
                }
                acc[0][nt] = wmma_bf16(afrag[0], bq[nt], acc[0][nt]);
                acc[1][nt] = wmma_bf16(afrag[1], bq[nt], acc[1][nt]);
            }
        }
    };

#if USE_ASYNC
    async_cp_row64(asrc, &As[0][lrow * LDA + lkoff]);
    int cur = 0;
    for (int k0 = 0; k0 < I_DIM; k0 += KB) {
        wait_async0();
        __syncthreads();
        if (k0 + KB < I_DIM) {
            async_cp_row64(asrc + k0 + KB, &As[cur ^ 1][lrow * LDA + lkoff]);
#pragma unroll
            for (int nt = 0; nt < 4; ++nt) {
                const long ncol = nbase + wn * 64 + nt * 16 + (lane & 15);
                __builtin_prefetch(w2m + wbase + ncol * I_DIM + k0 + KB, 0, 0);
            }
        }
        compute(As[cur], k0);
        cur ^= 1;
    }
#else
    for (int k0 = 0; k0 < I_DIM; k0 += KB) {
        __syncthreads();
        const uint4* s = (const uint4*)(asrc + k0);
        uint4* d = (uint4*)&As[0][lrow * LDA + lkoff];
        d[0] = s[0]; d[1] = s[1]; d[2] = s[2]; d[3] = s[3];
        __syncthreads();
        compute(As[0], k0);
    }
#endif

    // epilogue: weighted scatter-add to out[token]
    int   toks[2][8];
    float wgts[2][8];
#pragma unroll
    for (int mt = 0; mt < 2; ++mt)
#pragma unroll
        for (int j = 0; j < 8; ++j) {
            int mloc = (lane < 16) ? j : (8 + j);
            int row = row0 + wm * 32 + mt * 16 + mloc;
            toks[mt][j] = g_token[row];
            wgts[mt][j] = g_weight[row];
        }
#pragma unroll
    for (int mt = 0; mt < 2; ++mt) {
#pragma unroll
        for (int nt = 0; nt < 4; ++nt) {
            const int col = nbase + wn * 64 + nt * 16 + (lane & 15);
#pragma unroll
            for (int j = 0; j < 8; ++j) {
                if (toks[mt][j] >= 0) {
                    atomicAdd(&out[(long)toks[mt][j] * H_DIM + col],
                              wgts[mt][j] * acc[mt][nt][j]);
                }
            }
        }
    }
}

// ---------------- host launch ----------------
extern "C" void kernel_launch(void* const* d_in, const int* in_sizes, int n_in,
                              void* d_out, int out_size, void* d_ws, size_t ws_size,
                              hipStream_t stream) {
    const float* x    = (const float*)d_in[0];
    const float* tw   = (const float*)d_in[1];
    const int*   tids = (const int*)  d_in[2];
    const float* w13  = (const float*)d_in[3];
    const float* w2   = (const float*)d_in[4];
    const float* a1   = (const float*)d_in[5];
    const float* b1   = (const float*)d_in[6];
    const float* a2   = (const float*)d_in[7];
    const float* b2   = (const float*)d_in[8];
    float* out = (float*)d_out;

    // workspace carve-up (256B aligned)
    char* ws = (char*)d_ws;
    size_t off = 0;
    auto carve = [&](size_t bytes) { char* p = ws + off; off = (off + bytes + 255) & ~(size_t)255; return p; };
    __bf16* xb    = (__bf16*)carve((size_t)T_TOK * H_DIM * 2);                 // 64 MB
    __bf16* w13m  = (__bf16*)carve((size_t)E_NUM * 2 * I_DIM * H_DIM * 2);     // 64 MB
    __bf16* w2m   = (__bf16*)carve((size_t)E_NUM * H_DIM * I_DIM * 2);         // 32 MB
    __bf16* actws = (__bf16*)carve((size_t)ROWS_CAP * I_DIM * 2);              // 66 MB
    int*    g_token  = (int*)  carve((size_t)ROWS_CAP * 4);
    float*  g_weight = (float*)carve((size_t)ROWS_CAP * 4);
    int*    meta     = (int*)  carve((size_t)(32 + 2 * MAX_TILES) * 4);
    (void)ws_size; (void)in_sizes; (void)n_in; (void)out_size;

    const int B = 256;
    // 1) init / conversions / LoRA weight merge
    k_zero_out <<<(T_TOK * (long)H_DIM / 4 + B - 1) / B, B, 0, stream>>>((float4*)out, (long)T_TOK * H_DIM / 4);
    k_convert_x<<<(T_TOK * (long)H_DIM / 4 + B - 1) / B, B, 0, stream>>>(x, xb, (long)T_TOK * H_DIM / 4);
    k_merge_w13<<<((long)E_NUM * 2 * I_DIM * H_DIM + B - 1) / B, B, 0, stream>>>(w13, a1, b1, w13m);
    k_merge_w2 <<<((long)E_NUM * H_DIM * I_DIM + B - 1) / B, B, 0, stream>>>(w2, a2, b2, w2m);
    // 2) routing: fill -> count -> scan -> scatter
    k_fill   <<<(ROWS_CAP + B - 1) / B, B, 0, stream>>>(g_token, g_weight, meta);
    k_count  <<<(T_TOK + B - 1) / B, B, 0, stream>>>(tids, meta);
    k_scan   <<<1, 1, 0, stream>>>(meta);
    k_scatter<<<(T_TOK + B - 1) / B, B, 0, stream>>>(tids, tw, g_token, g_weight, meta);
    // 3) grouped GEMMs
    k_gemm1<<<dim3(MAX_TILES, I_DIM / 128), B, 0, stream>>>(xb, w13m, actws, g_token, meta);
    k_gemm2<<<dim3(MAX_TILES, H_DIM / 128), B, 0, stream>>>(actws, w2m, out, g_token, g_weight, meta);
}
